// SingleHeadLongformerModel_6665789243693
// MI455X (gfx1250) — matrix operand order repaired and checked by reading, not compiled, not run on hardware
//
#include <hip/hip_runtime.h>

typedef __attribute__((ext_vector_type(16))) _Float16 v16h;
typedef __attribute__((ext_vector_type(8)))  _Float16 v8h;
typedef __attribute__((ext_vector_type(8)))  float    v8f;

#define SEQ  2048
#define HSZ  768
#define NH   12
#define DH   64
#define FFN_ 3072
#define NL   12
#define WIN_ 256

#define WMMA(a,b,c) __builtin_amdgcn_wmma_f32_16x16x32_f16(false,(a),false,(b),(short)0,(c),false,false)

// A-matrix fragment (16x32 f16): lane holds row (lane&15); lane-half selects
// K = {16h..16h+15}: two contiguous 16B runs.
static __device__ __forceinline__ v16h ld_afrag(const _Float16* row, int half) {
  v8h lo = *(const v8h*)(row + 16 * half);
  v8h hi = *(const v8h*)(row + 16 * half + 8);
  return __builtin_shufflevector(lo, hi, 0,1,2,3,4,5,6,7,8,9,10,11,12,13,14,15);
}

// gfx1250 async copy global->LDS, 16B per lane, tracked by ASYNCcnt.
static __device__ __forceinline__ unsigned lds_off(const void* p) {
  return (unsigned)(size_t)p;            // generic LDS addr: low 32 bits = LDS offset
}
static __device__ __forceinline__ void async_cp16(unsigned lds, const void* g) {
  unsigned long long ga = (unsigned long long)(size_t)g;
  asm volatile("global_load_async_to_lds_b128 %0, %1, off"
               :: "v"(lds), "v"(ga) : "memory");
}
#define ASYNC_WAIT(n) asm volatile("s_wait_asynccnt " #n ::: "memory")

// ---------------------------------------------------------------------------
// Weight f32 -> f16 with transpose:  WT[n][k] = (f16) W[k][n]
// ---------------------------------------------------------------------------
__global__ __launch_bounds__(256) void convT_kernel(
    const float* __restrict__ W, _Float16* __restrict__ WT, int K, int N)
{
  __shared__ float t[64][65];
  const int k0 = blockIdx.x * 64, n0 = blockIdx.y * 64, tid = threadIdx.x;
  #pragma unroll
  for (int i = 0; i < 16; i++) {
    int e = tid + i * 256, r = e >> 6, c = e & 63;
    t[r][c] = W[(size_t)(k0 + r) * N + n0 + c];          // coalesced read
  }
  __syncthreads();
  #pragma unroll
  for (int i = 0; i < 16; i++) {
    int e = tid + i * 256, r = e >> 6, c = e & 63;       // r: n, c: k
    WT[(size_t)(n0 + r) * K + k0 + c] = (_Float16)t[c][r];  // coalesced write
  }
}

// ---------------------------------------------------------------------------
// WMMA GEMM: out = act(A_f16[M,K] @ WT_f16[N,K]^T + bias) * s
// 128x64 tile, 8 waves, K-step 32, double-buffered async LDS staging.
// ---------------------------------------------------------------------------
__global__ __launch_bounds__(256) void gemm_kernel(
    const _Float16* __restrict__ A, const _Float16* __restrict__ WT,
    const float* __restrict__ bias, float* __restrict__ outF,
    _Float16* __restrict__ outH, int M, int K, int N, int act, float scaleOut,
    int htrans)
{
  __shared__ __align__(32) _Float16 sA[2][128 * 32];   // [row][k]
  __shared__ __align__(32) _Float16 sBT[2][64 * 32];   // [n][k]

  const int tid  = threadIdx.x;
  const int lane = tid & 31, w = tid >> 5;
  const int half = lane >> 4, l16 = lane & 15;
  const int rowBase = blockIdx.y * 128;
  const int colBase = blockIdx.x * 64;
  const int nk = K / 32;

  auto stage = [&](int kt, int b) {
    const int k0 = kt * 32;
    #pragma unroll
    for (int i = 0; i < 2; i++) {                        // A tile: 2 x b128
      int e = (tid + i * 256) * 8, r = e >> 5, c = e & 31;
      async_cp16(lds_off(&sA[b][e]), A + (size_t)(rowBase + r) * K + k0 + c);
    }
    {                                                    // B tile: 1 x b128
      int e = tid * 8, n = e >> 5, c = e & 31;
      async_cp16(lds_off(&sBT[b][e]), WT + (size_t)(colBase + n) * K + k0 + c);
    }
  };

  v8f acc[4];
  for (int i = 0; i < 4; i++) acc[i] = {};

  stage(0, 0);
  for (int kt = 0; kt < nk; kt++) {
    const int b = kt & 1;
    if (kt + 1 < nk) { stage(kt + 1, b ^ 1); ASYNC_WAIT(0x3); }
    else             { ASYNC_WAIT(0x0); }
    __syncthreads();

    v16h a = ld_afrag(&sA[b][(w * 16 + l16) * 32], half);
    #pragma unroll
    for (int nt = 0; nt < 4; nt++) {
      v16h bb = *(const v16h*)(&sBT[b][(nt * 16 + l16) * 32 + 16 * half]);
      acc[nt] = WMMA(a, bb, acc[nt]);
    }
    __syncthreads();
  }

  #pragma unroll
  for (int nt = 0; nt < 4; nt++) {
    #pragma unroll
    for (int r = 0; r < 8; r++) {
      int row = rowBase + w * 16 + r + 8 * half;
      int col = colBase + nt * 16 + l16;
      float v = acc[nt][r] + bias[col];
      if (act == 1) v = 0.5f * v * (1.0f + erff(v * 0.70710678118f));
      v *= scaleOut;
      if (outF) outF[(size_t)row * N + col] = v;
      if (outH) {
        if (htrans) outH[(size_t)col * M + row] = (_Float16)v;   // [N][M]
        else        outH[(size_t)row * N + col] = (_Float16)v;   // [M][N]
      }
    }
  }
}

// ---------------------------------------------------------------------------
// Sliding-window flash attention. grid = (qtile=2, chunk=8, head=12).
// 8 waves x 16 query rows; key tiles of 32, double-buffered async staging;
// fully-padded tiles skipped (validity is 256-aligned -> uniform per tile).
// V comes in pre-transposed [d][s].
// ---------------------------------------------------------------------------
__global__ __launch_bounds__(256) void attn_kernel(
    const _Float16* __restrict__ qh, const _Float16* __restrict__ kh,
    const _Float16* __restrict__ vT, const int* __restrict__ km,
    _Float16* __restrict__ ah)
{
  __shared__ __align__(32) _Float16 sQ[128 * 64];
  __shared__ __align__(32) _Float16 sK[2][32 * 64];    // [key][d]
  __shared__ __align__(32) _Float16 sVT[2][64 * 32];   // [d][key]
  __shared__ float sS[8][16 * 32];
  __shared__ __align__(32) _Float16 sP[8][16 * 32];
  __shared__ float sM[8][16], sL[8][16], sC[8][16];

  const int tid = threadIdx.x, lane = tid & 31, w = tid >> 5;
  const int half = lane >> 4, l16 = lane & 15;
  const int qt = blockIdx.x, ch = blockIdx.y, hd = blockIdx.z;
  const int qBase = ch * WIN_ + qt * 128;
  const int hcol  = hd * DH;
  const int gBase = ch * WIN_ - WIN_;
  const int tlo = (ch == 0) ? 8 : 0;                   // skip left padding
  const int thi = (ch == SEQ / WIN_ - 1) ? 16 : 24;    // skip right padding

  auto stageKV = [&](int t, int b) {
    const int g0 = gBase + t * 32;
    {                                                  // K tile: 1 x b128
      int e = tid * 8, ky = e >> 6, c = e & 63;
      async_cp16(lds_off(&sK[b][e]), kh + (size_t)(g0 + ky) * HSZ + hcol + c);
    }
    {                                                  // V^T tile: 1 x b128
      int e = tid * 8, d = e >> 5, ky = e & 31;
      async_cp16(lds_off(&sVT[b][e]), vT + (size_t)(hcol + d) * SEQ + g0 + ky);
    }
  };

  // stage Q tile 128x64: 4 x b128
  #pragma unroll
  for (int i = 0; i < 4; i++) {
    int e = (tid + i * 256) * 8, r = e >> 6, c = e & 63;
    async_cp16(lds_off(&sQ[e]), qh + (size_t)(qBase + r) * HSZ + hcol + c);
  }
  stageKV(tlo, 0);
  if (lane < 16) { sM[w][lane] = -1e30f; sL[w][lane] = 0.f; }
  ASYNC_WAIT(0x2);                                     // Q done, first KV may fly
  __syncthreads();

  v16h a0 = ld_afrag(sQ + (w * 16 + l16) * 64, half);
  v16h a1 = ld_afrag(sQ + (w * 16 + l16) * 64 + 32, half);
  v8f o[4];
  for (int i = 0; i < 4; i++) o[i] = {};

  for (int t = tlo; t < thi; t++) {
    const int b = (t - tlo) & 1;
    if (t + 1 < thi) { stageKV(t + 1, b ^ 1); ASYNC_WAIT(0x2); }
    else             { ASYNC_WAIT(0x0); }
    __syncthreads();

    // S(16x32) = Q(16x64) @ K^T
    #pragma unroll
    for (int nt = 0; nt < 2; nt++) {
      v8f s = {};
      v16h b0 = *(const v16h*)(&sK[b][(nt * 16 + l16) * 64 + 16 * half]);
      v16h b1 = *(const v16h*)(&sK[b][(nt * 16 + l16) * 64 + 32 + 16 * half]);
      s = WMMA(a0, b0, s);
      s = WMMA(a1, b1, s);
      #pragma unroll
      for (int r = 0; r < 8; r++) {
        int iq = qt * 128 + w * 16 + r + 8 * half;
        int kk = t * 32 + nt * 16 + l16;
        bool ok = (kk >= iq) && (kk <= iq + 2 * WIN_) && (km[gBase + kk] != 0);
        sS[w][(r + 8 * half) * 32 + nt * 16 + l16] = ok ? s[r] : -1e9f;
      }
    }
    __syncthreads();

    // online softmax: lanes 0..15 own the wave's 16 rows
    if (lane < 16) {
      float mOld = sM[w][lane], lOld = sL[w][lane];
      float sv[32], mx = -1e30f;
      #pragma unroll
      for (int j = 0; j < 32; j++) { sv[j] = sS[w][lane * 32 + j]; mx = fmaxf(mx, sv[j]); }
      float mNew = fmaxf(mOld, mx);
      float c = __expf(mOld - mNew);
      float sum = 0.f;
      #pragma unroll
      for (int j = 0; j < 32; j++) {
        float p = __expf(sv[j] - mNew);
        sP[w][lane * 32 + j] = (_Float16)p;
        sum += p;
      }
      sM[w][lane] = mNew; sL[w][lane] = lOld * c + sum; sC[w][lane] = c;
    }
    __syncthreads();

    // rescale O and accumulate P @ V
    float cr[8];
    #pragma unroll
    for (int r = 0; r < 8; r++) cr[r] = sC[w][r + 8 * half];
    v16h ap = ld_afrag(&sP[w][l16 * 32], half);
    #pragma unroll
    for (int nt = 0; nt < 4; nt++) {
      #pragma unroll
      for (int r = 0; r < 8; r++) o[nt][r] *= cr[r];
      v16h bv = *(const v16h*)(&sVT[b][(nt * 16 + l16) * 32 + 16 * half]);
      o[nt] = WMMA(ap, bv, o[nt]);
    }
    __syncthreads();
  }

  float lr[8];
  #pragma unroll
  for (int r = 0; r < 8; r++) {
    float l = sL[w][r + 8 * half];
    lr[r] = (l > 0.f) ? 1.0f / l : 0.f;
  }
  #pragma unroll
  for (int nt = 0; nt < 4; nt++)
    #pragma unroll
    for (int r = 0; r < 8; r++) {
      int srow = qBase + w * 16 + r + 8 * half;
      ah[(size_t)srow * HSZ + hcol + nt * 16 + l16] = (_Float16)(o[nt][r] * lr[r]);
    }
}

// ---------------------------------------------------------------------------
// Embedding gather + LayerNorm / residual + LayerNorm (one block per token)
// ---------------------------------------------------------------------------
__global__ __launch_bounds__(256) void embed_ln_kernel(
    const int* __restrict__ ids, const float* __restrict__ we,
    const float* __restrict__ pe, const float* __restrict__ sc,
    const float* __restrict__ bi, float* __restrict__ x, _Float16* __restrict__ xh)
{
  __shared__ float red[256];
  __shared__ float stats[2];
  int s = blockIdx.x, tid = threadIdx.x;
  int tok = ids[s];
  float v[3], sum = 0.f;
  #pragma unroll
  for (int i = 0; i < 3; i++) {
    int hh = tid + i * 256;
    v[i] = we[(size_t)tok * HSZ + hh] + pe[(size_t)s * HSZ + hh];
    sum += v[i];
  }
  red[tid] = sum; __syncthreads();
  for (int st = 128; st > 0; st >>= 1) { if (tid < st) red[tid] += red[tid + st]; __syncthreads(); }
  if (tid == 0) stats[0] = red[0] / HSZ;
  __syncthreads();
  float mu = stats[0], vs = 0.f;
  #pragma unroll
  for (int i = 0; i < 3; i++) { float d = v[i] - mu; vs += d * d; }
  red[tid] = vs; __syncthreads();
  for (int st = 128; st > 0; st >>= 1) { if (tid < st) red[tid] += red[tid + st]; __syncthreads(); }
  if (tid == 0) stats[1] = red[0] / HSZ;
  __syncthreads();
  float inv = rsqrtf(stats[1] + 1e-5f);
  #pragma unroll
  for (int i = 0; i < 3; i++) {
    int hh = tid + i * 256;
    float y = (v[i] - mu) * inv * sc[hh] + bi[hh];
    x[(size_t)s * HSZ + hh] = y;
    xh[(size_t)s * HSZ + hh] = (_Float16)y;
  }
}

__global__ __launch_bounds__(256) void add_ln_kernel(
    const float* __restrict__ xin, const float* __restrict__ p,
    const float* __restrict__ sc, const float* __restrict__ bi,
    float* __restrict__ xout, _Float16* __restrict__ xh)
{
  __shared__ float red[256];
  __shared__ float stats[2];
  int s = blockIdx.x, tid = threadIdx.x;
  float v[3], sum = 0.f;
  #pragma unroll
  for (int i = 0; i < 3; i++) {
    int hh = tid + i * 256;
    v[i] = xin[(size_t)s * HSZ + hh] + p[(size_t)s * HSZ + hh];
    sum += v[i];
  }
  red[tid] = sum; __syncthreads();
  for (int st = 128; st > 0; st >>= 1) { if (tid < st) red[tid] += red[tid + st]; __syncthreads(); }
  if (tid == 0) stats[0] = red[0] / HSZ;
  __syncthreads();
  float mu = stats[0], vs = 0.f;
  #pragma unroll
  for (int i = 0; i < 3; i++) { float d = v[i] - mu; vs += d * d; }
  red[tid] = vs; __syncthreads();
  for (int st = 128; st > 0; st >>= 1) { if (tid < st) red[tid] += red[tid + st]; __syncthreads(); }
  if (tid == 0) stats[1] = red[0] / HSZ;
  __syncthreads();
  float inv = rsqrtf(stats[1] + 1e-5f);
  #pragma unroll
  for (int i = 0; i < 3; i++) {
    int hh = tid + i * 256;
    float y = (v[i] - mu) * inv * sc[hh] + bi[hh];
    xout[(size_t)s * HSZ + hh] = y;
    xh[(size_t)s * HSZ + hh] = (_Float16)y;
  }
}

// classifier head (tiny)
__global__ __launch_bounds__(256) void head1_kernel(
    const float* __restrict__ x, const float* __restrict__ W,
    const float* __restrict__ b, float* __restrict__ h)
{
  int j = blockIdx.x * 256 + threadIdx.x;
  float acc = b[j];
  for (int i = 0; i < HSZ; i++) acc += x[i] * W[(size_t)i * 512 + j];
  h[j] = fmaxf(acc, 0.f);
}

__global__ __launch_bounds__(64) void head2_kernel(
    const float* __restrict__ h, const float* __restrict__ W,
    const float* __restrict__ b, float* __restrict__ out)
{
  int j = threadIdx.x;
  if (j >= 43) return;
  float acc = b[j];
  for (int i = 0; i < 512; i++) acc += h[i] * W[(size_t)i * 43 + j];
  out[j] = acc;
}

// ---------------------------------------------------------------------------
extern "C" void kernel_launch(void* const* d_in, const int* in_sizes, int n_in,
                              void* d_out, int out_size, void* d_ws, size_t ws_size,
                              hipStream_t stream)
{
  (void)in_sizes; (void)n_in; (void)out_size; (void)ws_size;
  const int*   ids  = (const int*)d_in[0];
  const int*   am   = (const int*)d_in[1];
  const float* we   = (const float*)d_in[2];
  const float* pe   = (const float*)d_in[3];
  const float* elns = (const float*)d_in[4];
  const float* elnb = (const float*)d_in[5];
  const float* Wq = (const float*)d_in[6];   const float* bq = (const float*)d_in[7];
  const float* Wk = (const float*)d_in[8];   const float* bk = (const float*)d_in[9];
  const float* Wv = (const float*)d_in[10];  const float* bv = (const float*)d_in[11];
  const float* Wo = (const float*)d_in[12];  const float* bo = (const float*)d_in[13];
  const float* ln1s = (const float*)d_in[14]; const float* ln1b = (const float*)d_in[15];
  const float* W1 = (const float*)d_in[16];  const float* b1 = (const float*)d_in[17];
  const float* W2 = (const float*)d_in[18];  const float* b2 = (const float*)d_in[19];
  const float* ln2s = (const float*)d_in[20]; const float* ln2b = (const float*)d_in[21];
  const float* hW1 = (const float*)d_in[22]; const float* hb1 = (const float*)d_in[23];
  const float* hW2 = (const float*)d_in[24]; const float* hb2 = (const float*)d_in[25];
  float* out = (float*)d_out;

  char* ws = (char*)d_ws;
  size_t off = 0;
  auto alloc = [&](size_t bytes) -> void* {
    void* p = ws + off;
    off = (off + bytes + 255) & ~(size_t)255;
    return p;
  };
  float*    x    = (float*)   alloc((size_t)SEQ * HSZ * 4);
  float*    proj = (float*)   alloc((size_t)SEQ * HSZ * 4);
  _Float16* xh   = (_Float16*)alloc((size_t)SEQ * HSZ * 2);
  _Float16* qhb  = (_Float16*)alloc((size_t)SEQ * HSZ * 2);
  _Float16* khb  = (_Float16*)alloc((size_t)SEQ * HSZ * 2);
  _Float16* vTb  = (_Float16*)alloc((size_t)SEQ * HSZ * 2);   // [d][s]
  _Float16* ahb  = (_Float16*)alloc((size_t)SEQ * HSZ * 2);
  _Float16* hhb  = (_Float16*)alloc((size_t)SEQ * FFN_ * 2);
  _Float16* wT   = (_Float16*)alloc((size_t)HSZ * FFN_ * 2);  // per-GEMM weight^T f16
  float*    h512 = (float*)   alloc(512 * 4);

  const dim3 blk(256);
  const dim3 gH (HSZ / 64,  SEQ / 128);
  const dim3 gF1(FFN_ / 64, SEQ / 128);
  const dim3 gAt(2, SEQ / WIN_, NH);
  const dim3 cHH(HSZ / 64, HSZ / 64);     // convT grids: (K/64, N/64)
  const dim3 cHF(HSZ / 64, FFN_ / 64);
  const dim3 cFH(FFN_ / 64, HSZ / 64);

  embed_ln_kernel<<<SEQ, blk, 0, stream>>>(ids, we, pe, elns, elnb, x, xh);

  for (int l = 0; l < NL; l++) {
    const float* wq = Wq + (size_t)l * HSZ * HSZ;  const float* bql = bq + (size_t)l * HSZ;
    const float* wk = Wk + (size_t)l * HSZ * HSZ;  const float* bkl = bk + (size_t)l * HSZ;
    const float* wv = Wv + (size_t)l * HSZ * HSZ;  const float* bvl = bv + (size_t)l * HSZ;
    const float* wo = Wo + (size_t)l * HSZ * HSZ;  const float* bol = bo + (size_t)l * HSZ;
    const float* w1 = W1 + (size_t)l * HSZ * FFN_; const float* b1l = b1 + (size_t)l * FFN_;
    const float* w2 = W2 + (size_t)l * FFN_ * HSZ; const float* b2l = b2 + (size_t)l * HSZ;

    convT_kernel<<<cHH, blk, 0, stream>>>(wq, wT, HSZ, HSZ);
    gemm_kernel<<<gH, blk, 0, stream>>>(xh, wT, bql, nullptr, qhb, SEQ, HSZ, HSZ, 0, 0.125f, 0);
    convT_kernel<<<cHH, blk, 0, stream>>>(wk, wT, HSZ, HSZ);
    gemm_kernel<<<gH, blk, 0, stream>>>(xh, wT, bkl, nullptr, khb, SEQ, HSZ, HSZ, 0, 1.0f, 0);
    convT_kernel<<<cHH, blk, 0, stream>>>(wv, wT, HSZ, HSZ);
    gemm_kernel<<<gH, blk, 0, stream>>>(xh, wT, bvl, nullptr, vTb, SEQ, HSZ, HSZ, 0, 1.0f, 1);

    attn_kernel<<<gAt, blk, 0, stream>>>(qhb, khb, vTb, am, ahb);

    convT_kernel<<<cHH, blk, 0, stream>>>(wo, wT, HSZ, HSZ);
    gemm_kernel<<<gH, blk, 0, stream>>>(ahb, wT, bol, proj, nullptr, SEQ, HSZ, HSZ, 0, 1.0f, 0);
    add_ln_kernel<<<SEQ, blk, 0, stream>>>(x, proj, ln1s + (size_t)l * HSZ, ln1b + (size_t)l * HSZ, x, xh);

    convT_kernel<<<cHF, blk, 0, stream>>>(w1, wT, HSZ, FFN_);
    gemm_kernel<<<gF1, blk, 0, stream>>>(xh, wT, b1l, nullptr, hhb, SEQ, HSZ, FFN_, 1, 1.0f, 0);
    convT_kernel<<<cFH, blk, 0, stream>>>(w2, wT, FFN_, HSZ);
    gemm_kernel<<<gH, blk, 0, stream>>>(hhb, wT, b2l, proj, nullptr, SEQ, FFN_, HSZ, 0, 1.0f, 0);
    add_ln_kernel<<<SEQ, blk, 0, stream>>>(x, proj, ln2s + (size_t)l * HSZ, ln2b + (size_t)l * HSZ, x, xh);
  }

  head1_kernel<<<2, blk, 0, stream>>>(x, hW1, hb1, h512);
  head2_kernel<<<1, 64, 0, stream>>>(h512, hW2, hb2, out);
}
